// GCNGraphClassifier_50397146251362
// MI455X (gfx1250) — matrix operand order repaired and checked by reading, not compiled, not run on hardware
//
#include <hip/hip_runtime.h>
#include <hip/hip_bf16.h>

// GCN graph classifier for MI455X (gfx1250, wave32).
// Dense transforms: v_wmma_f32_16x16x32_bf16, bf16 operands pre-packed so the
// inner loop is pure global_load_b128 + wmma (no predication, no converts).
// Edge aggregation: L2-resident scatter with native f32 global atomics.

typedef __attribute__((ext_vector_type(16))) __bf16 v16bf;
typedef __attribute__((ext_vector_type(8)))  __bf16 v8bf;
typedef __attribute__((ext_vector_type(8)))  float  v8f;

static __device__ __forceinline__ __bf16 f2bf(float x) { return (__bf16)x; }

// ---------------- elementwise / setup kernels ----------------

__global__ void k_zero(float* __restrict__ p, long long n) {
  long long i = (long long)blockIdx.x * blockDim.x + threadIdx.x;
  if (i < n) p[i] = 0.0f;
}

__global__ void k_cvt_bf16(const float* __restrict__ in, __bf16* __restrict__ out,
                           long long n) {
  long long i = (long long)blockIdx.x * blockDim.x + threadIdx.x;
  if (i < n) out[i] = f2bf(in[i]);
}

__global__ void k_init_deg(float* __restrict__ deg, int n) {
  int i = blockIdx.x * blockDim.x + threadIdx.x;
  if (i < n) deg[i] = 1.0f;   // self-loop contribution
}

__global__ void k_edge_deg(const int* __restrict__ dst, float* __restrict__ deg, int E) {
  int e = blockIdx.x * blockDim.x + threadIdx.x;
  if (e < E) unsafeAtomicAdd(&deg[dst[e]], 1.0f);
}

__global__ void k_rsqrt(float* __restrict__ d, int n) {
  int i = blockIdx.x * blockDim.x + threadIdx.x;
  if (i < n) { float v = d[i]; d[i] = (v > 0.0f) ? rsqrtf(v) : 0.0f; }
}

// Transpose + convert weights: Wt[c*K + k] = bf16(W[k*Kout + c]).
// Wt is padded to KoutPad columns (zero-filled beforehand) so GEMM B-loads
// need no bounds check.
__global__ void k_wt_bf16(const float* __restrict__ W, __bf16* __restrict__ Wt,
                          int K, int Kout) {
  int i = blockIdx.x * blockDim.x + threadIdx.x;
  if (i < K * Kout) {
    int k = i % K, c = i / K;
    Wt[i] = f2bf(W[k * Kout + c]);
  }
}

// ---------------- WMMA GEMM: Out[N,Kout] = A[N,K] @ W[K,Kout] ----------------
// One wave computes a 16(M) x 64(N) tile with 4 accumulators; K is a
// compile-time constant so the k-loop fully unrolls.
// mode: 0 = plain store, 1 = +bias then ReLU, 2 = +bias.
template <int K>
__global__ void k_wmma_gemm(const __bf16* __restrict__ A, const __bf16* __restrict__ Wt,
                            const float* __restrict__ bias, float* __restrict__ Out,
                            int N, int Kout, int nTiles, int mode) {
  int gid  = blockIdx.x * blockDim.x + threadIdx.x;
  int wid  = gid >> 5;                 // wave-uniform
  int lane = threadIdx.x & 31;
  int mTiles = (N + 15) >> 4;
  if (wid >= mTiles * nTiles) return;  // wave-uniform exit: EXEC stays full for WMMA
  int mT = wid / nTiles;
  int nT = wid - mT * nTiles;
  int half = lane >> 4;
  int l15  = lane & 15;
  int row  = mT * 16 + l15;
  int rowc = row < N ? row : N - 1;    // clamp: loads unconditional, stores masked

  const __bf16* aRow = A + (long long)rowc * K;
  const __bf16* bCol0 = Wt + (long long)(nT * 64 + l15) * K + 8 * half;

  v8f acc[4] = {};

  #pragma unroll
  for (int k0 = 0; k0 < K; k0 += 32) {
    // A fragment (16x32 bf16), ISA layout: lanes 0-15 K=[0..7]+[16..23],
    // lanes 16-31 K=[8..15]+[24..31]; row M = lane&15. Two b128 loads.
    const __bf16* ap = aRow + k0 + 8 * half;
    v8bf a_lo = *(const v8bf*)(ap);
    v8bf a_hi = *(const v8bf*)(ap + 16);
    v16bf a = __builtin_shufflevector(a_lo, a_hi,
        0, 1, 2, 3, 4, 5, 6, 7, 8, 9, 10, 11, 12, 13, 14, 15);
    #pragma unroll
    for (int s = 0; s < 4; ++s) {
      // B fragment (32x16 bf16): col = lane&15; lanes 0-15 K=[0..7]+[16..23],
      // lanes 16-31 K=[8..15]+[24..31] (same order as A, from transposed, padded Wt).
      const __bf16* bp = bCol0 + (long long)s * 16 * K + k0;
      v8bf b_lo = *(const v8bf*)(bp);
      v8bf b_hi = *(const v8bf*)(bp + 16);
      v16bf b = __builtin_shufflevector(b_lo, b_hi,
          0, 1, 2, 3, 4, 5, 6, 7, 8, 9, 10, 11, 12, 13, 14, 15);
      acc[s] = __builtin_amdgcn_wmma_f32_16x16x32_bf16(
          false, a, false, b, (short)0, acc[s], false, false);
    }
  }

  // C/D layout: VGPR v -> row = mT*16 + 8*half + v, col = lane&15.
  #pragma unroll
  for (int s = 0; s < 4; ++s) {
    int col = nT * 64 + s * 16 + l15;
    if (col >= Kout) continue;
    float bv = (mode != 0) ? bias[col] : 0.0f;
    #pragma unroll
    for (int v = 0; v < 8; ++v) {
      int r = mT * 16 + half * 8 + v;
      if (r < N) {
        float val = acc[s][v] + bv;
        if (mode == 1) val = fmaxf(val, 0.0f);
        Out[(long long)r * Kout + col] = val;
      }
    }
  }
}

// ---------------- edge scatter: agg[dst] += t[src] * d[src]*d[dst] ----------------
// Wave per edge; b128 gathers, per-float native atomics (one wave covers a
// full contiguous feature row -> coalesced L2 transactions).
__global__ void k_scatter(const float* __restrict__ t, const int* __restrict__ src,
                          const int* __restrict__ dst, const float* __restrict__ dis,
                          float* __restrict__ agg, int E, int C) {
  int gid  = blockIdx.x * blockDim.x + threadIdx.x;
  int w    = gid >> 5;
  int lane = gid & 31;
  if (w >= E) return;
  int s = src[w], d = dst[w];
  float en = dis[s] * dis[d];
  const float* tp = t + (long long)s * C;
  float* ap = agg + (long long)d * C;
  for (int c = 4 * lane; c < C; c += 128) {
    float4 v = *(const float4*)(tp + c);
    unsafeAtomicAdd(&ap[c + 0], v.x * en);
    unsafeAtomicAdd(&ap[c + 1], v.y * en);
    unsafeAtomicAdd(&ap[c + 2], v.z * en);
    unsafeAtomicAdd(&ap[c + 3], v.w * en);
  }
}

// h = agg + t * d[i]^2 + bias (self-loop term folded in), optional ReLU.
// If outbf != nullptr, write bf16 (activation feeding the next WMMA GEMM),
// else write f32 in place.
__global__ void k_combine(float* __restrict__ agg, const float* __restrict__ t,
                          const float* __restrict__ dis, const float* __restrict__ bias,
                          __bf16* __restrict__ outbf, int N, int C, int relu) {
  long long idx = (long long)blockIdx.x * blockDim.x + threadIdx.x;
  if (idx >= (long long)N * C) return;
  int i = (int)(idx / C);
  int c = (int)(idx - (long long)i * C);
  float ds = dis[i];
  float v = agg[idx] + t[idx] * ds * ds + bias[c];
  if (relu) v = fmaxf(v, 0.0f);
  if (outbf) outbf[idx] = f2bf(v);
  else       agg[idx] = v;
}

// ---------------- global mean pool ----------------
__global__ void k_pool(const float* __restrict__ h, const int* __restrict__ batch,
                       float* __restrict__ sums, float* __restrict__ cnts, int N, int C) {
  int gid  = blockIdx.x * blockDim.x + threadIdx.x;
  int w    = gid >> 5;
  int lane = gid & 31;
  if (w >= N) return;
  int g = batch[w];
  const float* hp = h + (long long)w * C;
  float* sp = sums + (long long)g * C;
  for (int c = 4 * lane; c < C; c += 128) {
    float4 v = *(const float4*)(hp + c);
    unsafeAtomicAdd(&sp[c + 0], v.x);
    unsafeAtomicAdd(&sp[c + 1], v.y);
    unsafeAtomicAdd(&sp[c + 2], v.z);
    unsafeAtomicAdd(&sp[c + 3], v.w);
  }
  if (lane == 0) unsafeAtomicAdd(&cnts[g], 1.0f);
}

__global__ void k_pool_div(const float* __restrict__ sums, const float* __restrict__ cnts,
                           float* __restrict__ hg, int G, int C) {
  int idx = blockIdx.x * blockDim.x + threadIdx.x;
  if (idx < G * C) hg[idx] = sums[idx] / fmaxf(cnts[idx / C], 1.0f);
}

// ---------------- launch ----------------

extern "C" void kernel_launch(void* const* d_in, const int* in_sizes, int n_in,
                              void* d_out, int out_size, void* d_ws, size_t ws_size,
                              hipStream_t stream) {
  const float* x    = (const float*)d_in[0];
  const int*   ei   = (const int*)d_in[1];
  const int*   batch= (const int*)d_in[2];
  const float* W1   = (const float*)d_in[3];  const float* b1   = (const float*)d_in[4];
  const float* W2   = (const float*)d_in[5];  const float* b2   = (const float*)d_in[6];
  const float* W3   = (const float*)d_in[7];  const float* b3   = (const float*)d_in[8];
  const float* fcW1 = (const float*)d_in[9];  const float* fcb1 = (const float*)d_in[10];
  const float* fcW2 = (const float*)d_in[11]; const float* fcb2 = (const float*)d_in[12];
  const float* fcW  = (const float*)d_in[13]; const float* fcb  = (const float*)d_in[14];

  const int N = in_sizes[0] / 128;
  const int E = in_sizes[1] / 2;
  const int G = 256;
  const int* srcI = ei;
  const int* dstI = ei + E;

  // workspace layout (float units; all offsets stay 16B aligned)
  float* ws = (float*)d_ws;
  size_t off = 0;
  float*  dis  = ws + off; off += (size_t)N;
  float*  hA   = ws + off; off += (size_t)N * 256;       // gemm output t = h@W
  float*  hB   = ws + off; off += (size_t)N * 256;       // aggregation / layer-3 output
  __bf16* xbf  = (__bf16*)(ws + off); off += (size_t)N * 64;   // N*128 bf16
  __bf16* hCbf = (__bf16*)(ws + off); off += (size_t)N * 64;   // N*128 bf16 activations
  float*  sums = ws + off; off += (size_t)G * 256;
  float*  cnts = ws + off; off += (size_t)G;             // contiguous after sums
  float*  hg   = ws + off; off += (size_t)G * 256;
  __bf16* hgbf = (__bf16*)(ws + off); off += (size_t)G * 128;  // G*256 bf16
  float*  zfc  = ws + off; off += (size_t)G * 128;
  __bf16* zbf  = (__bf16*)(ws + off); off += (size_t)G * 64;   // G*128 bf16
  float*  feat = ws + off; off += (size_t)G * 64;
  __bf16* fbf  = (__bf16*)(ws + off); off += (size_t)G * 32;   // G*64 bf16
  __bf16* Wt   = (__bf16*)(ws + off); off += (size_t)256 * 256 / 2; // padded bf16 weights
  (void)ws_size; (void)n_in; (void)out_size;

  const int BT = 256;
  auto cdiv = [](long long a, long long b) { return (int)((a + b - 1) / b); };

  auto gemm = [&](const __bf16* A, const float* W, const float* bias, float* Out,
                  int n, int k, int kout, int mode) {
    int kpad = ((kout + 63) / 64) * 64;
    // zero-fill padded Wt, then transpose-convert the live columns
    k_zero<<<cdiv((long long)kpad * k / 2, BT), BT, 0, stream>>>(
        (float*)Wt, (long long)kpad * k / 2);
    k_wt_bf16<<<cdiv((long long)k * kout, BT), BT, 0, stream>>>(W, Wt, k, kout);
    int mTiles = (n + 15) / 16, nTiles = kpad / 64;
    long long waves = (long long)mTiles * nTiles;
    int blocks = cdiv(waves * 32, BT);
    if (k == 64)
      k_wmma_gemm<64><<<blocks, BT, 0, stream>>>(A, Wt, bias, Out, n, kout, nTiles, mode);
    else if (k == 128)
      k_wmma_gemm<128><<<blocks, BT, 0, stream>>>(A, Wt, bias, Out, n, kout, nTiles, mode);
    else
      k_wmma_gemm<256><<<blocks, BT, 0, stream>>>(A, Wt, bias, Out, n, kout, nTiles, mode);
  };

  // degree / symmetric normalization (self-loop => deg init 1.0)
  k_init_deg<<<cdiv(N, BT), BT, 0, stream>>>(dis, N);
  k_edge_deg<<<cdiv(E, BT), BT, 0, stream>>>(dstI, dis, E);
  k_rsqrt  <<<cdiv(N, BT), BT, 0, stream>>>(dis, N);

  // input activations -> bf16 once
  k_cvt_bf16<<<cdiv((long long)N * 128, BT), BT, 0, stream>>>(x, xbf, (long long)N * 128);

  // ---- GCN layer 1: 128 -> 128, ReLU ----
  gemm(xbf, W1, nullptr, hA, N, 128, 128, 0);
  k_zero   <<<cdiv((long long)N * 128, BT), BT, 0, stream>>>(hB, (long long)N * 128);
  k_scatter<<<cdiv((long long)E * 32, BT), BT, 0, stream>>>(hA, srcI, dstI, dis, hB, E, 128);
  k_combine<<<cdiv((long long)N * 128, BT), BT, 0, stream>>>(hB, hA, dis, b1, hCbf, N, 128, 1);

  // ---- GCN layer 2: 128 -> 128, ReLU ----
  gemm(hCbf, W2, nullptr, hA, N, 128, 128, 0);
  k_zero   <<<cdiv((long long)N * 128, BT), BT, 0, stream>>>(hB, (long long)N * 128);
  k_scatter<<<cdiv((long long)E * 32, BT), BT, 0, stream>>>(hA, srcI, dstI, dis, hB, E, 128);
  k_combine<<<cdiv((long long)N * 128, BT), BT, 0, stream>>>(hB, hA, dis, b2, hCbf, N, 128, 1);

  // ---- GCN layer 3: 128 -> 256, linear ----
  gemm(hCbf, W3, nullptr, hA, N, 128, 256, 0);
  k_zero   <<<cdiv((long long)N * 256, BT), BT, 0, stream>>>(hB, (long long)N * 256);
  k_scatter<<<cdiv((long long)E * 32, BT), BT, 0, stream>>>(hA, srcI, dstI, dis, hB, E, 256);
  k_combine<<<cdiv((long long)N * 256, BT), BT, 0, stream>>>(hB, hA, dis, b3, nullptr, N, 256, 0);

  // ---- global mean pool ----
  k_zero    <<<cdiv((long long)G * 257, BT), BT, 0, stream>>>(sums, (long long)G * 257);
  k_pool    <<<cdiv((long long)N * 32, BT), BT, 0, stream>>>(hB, batch, sums, cnts, N, 256);
  k_pool_div<<<cdiv((long long)G * 256, BT), BT, 0, stream>>>(sums, cnts, hg, G, 256);
  k_cvt_bf16<<<cdiv((long long)G * 256, BT), BT, 0, stream>>>(hg, hgbf, (long long)G * 256);

  // ---- MLP head (WMMA, bias/ReLU fused) ----
  gemm(hgbf, fcW1, fcb1, zfc, G, 256, 128, 1);
  k_cvt_bf16<<<cdiv((long long)G * 128, BT), BT, 0, stream>>>(zfc, zbf, (long long)G * 128);
  gemm(zbf, fcW2, fcb2, feat, G, 128, 64, 1);
  k_cvt_bf16<<<cdiv((long long)G * 64, BT), BT, 0, stream>>>(feat, fbf, (long long)G * 64);
  gemm(fbf, fcW, fcb, (float*)d_out, G, 64, 10, 2);   // logits -> d_out[0 : 2560]

  // second tuple output: feat [256,64] after logits
  hipMemcpyAsync((float*)d_out + (size_t)G * 10, feat,
                 (size_t)G * 64 * sizeof(float), hipMemcpyDeviceToDevice, stream);
}
